// GenericLSTM_1563368096071
// MI455X (gfx1250) — matrix operand order repaired and checked by reading, not compiled, not run on hardware
//
#include <hip/hip_runtime.h>
#include <hip/hip_bf16.h>
#include <stdint.h>

// ---------------- types / constants ----------------
typedef __attribute__((ext_vector_type(16))) __bf16 v16bf;
typedef __attribute__((ext_vector_type(8)))  float  v8f;

#define T_STEPS 2048
#define DMODEL  256
#define HDIM    256
#define BB      16      // batch rows per block
#define WCOLS   128     // gate columns per WG (32 hidden units * 4 gates)
#define NBLK    4
#define NWGCOL  8

// ---------------- helpers ----------------
__device__ __forceinline__ uint16_t f2bf(float f) {
  union { __hip_bfloat16 b; uint16_t u; } c;
  c.b = __float2bfloat16(f);                 // native v_cvt on gfx1250
  return c.u;
}
__device__ __forceinline__ float bf2f(uint16_t h) {
  union { uint16_t u; __hip_bfloat16 b; } c; c.u = h;
  return __bfloat162float(c.b);
}
__device__ __forceinline__ float sigm(float x) { return 1.0f / (1.0f + __expf(-x)); }
// tanh(x) = 1 - 2/(e^{2x}+1): one v_exp_f32, exact +-1 limits, no inf/inf NaN
__device__ __forceinline__ float tanh_fast(float x) {
  float e = __expf(2.0f * x);
  return 1.0f - 2.0f / (e + 1.0f);
}

// A fragment: 16x32 bf16 tile (row-major, stride 256) contiguous K order, same as B swizzle:
// lane(0..15)=row M, half=lane>>4 selects K {0..15} vs {16..31}
__device__ __forceinline__ v16bf load_a(const uint16_t* t, int lane, int kt) {
  int m = lane & 15;
  int base = m * 256 + kt * 32 + ((lane >> 4) << 4);
  union { v16bf v; uint4 q[2]; } f;
  f.q[0] = *(const uint4*)(t + base);       // K j=0..7  (+16h)
  f.q[1] = *(const uint4*)(t + base + 8);   // K j=8..15 (+16h)
  return f.v;
}
// B fragment: weights pre-swizzled at startup so each lane reads 32 contiguous bytes
__device__ __forceinline__ v16bf load_b(const uint16_t* w, int lane, int kt, int nt) {
  int base = ((kt << 3) + nt) * 512 + lane * 16;
  union { v16bf v; uint4 q[2]; } f;
  f.q[0] = *(const uint4*)(w + base);
  f.q[1] = *(const uint4*)(w + base + 8);
  return f.v;
}
#define WMMA_BF16(a, b, c) \
  __builtin_amdgcn_wmma_f32_16x16x32_bf16(false, (a), false, (b), (short)0, (c), false, false)

// one inter-WG barrier per timestep (cluster fast-path + agent-scope atomic sync)
__device__ __forceinline__ void block_barrier(unsigned* cnt, unsigned target) {
  __builtin_amdgcn_fence(__ATOMIC_RELEASE, "agent");   // flush h stores to L2
  __syncthreads();
  __builtin_amdgcn_s_cluster_barrier();                // s_barrier_signal/-wait -3 (NOP if unclustered)
  if (threadIdx.x == 0) {
    __hip_atomic_fetch_add(cnt, 1u, __ATOMIC_RELAXED, __HIP_MEMORY_SCOPE_AGENT);
    while (__hip_atomic_load(cnt, __ATOMIC_RELAXED, __HIP_MEMORY_SCOPE_AGENT) < target)
      __builtin_amdgcn_s_sleep(2);
  }
  __syncthreads();
  __builtin_amdgcn_fence(__ATOMIC_ACQUIRE, "agent");   // global_inv: kill stale WGP$ lines
}

// ---------------- init kernel ----------------
__global__ void lstm_init(unsigned* cnt, uint16_t* h0buf, uint16_t* h1buf) {
  int i = blockIdx.x * blockDim.x + threadIdx.x;
  if (i < NBLK) cnt[i] = 0;
  const int n = 2 * NBLK * BB * HDIM;   // [parity][blk][row][H]
  for (int j = i; j < n; j += gridDim.x * blockDim.x) { h0buf[j] = 0; h1buf[j] = 0; }
}

// ---------------- persistent LSTM kernel ----------------
__global__ __launch_bounds__(256, 1)
void lstm_persistent(const float* __restrict__ x,   const float* __restrict__ Wxh,
                     const float* __restrict__ bxh, const float* __restrict__ Whh,
                     const float* __restrict__ bhh, float* __restrict__ out,
                     uint16_t* __restrict__ h0buf,  uint16_t* __restrict__ h1buf,
                     unsigned* __restrict__ cnt) {
  __shared__ uint16_t sWxh[256 * WCOLS];   // 64 KB, swizzled bf16 slice
  __shared__ uint16_t sWhh[256 * WCOLS];   // 64 KB
  __shared__ uint16_t sXA[BB * 256];       // x_t tile, bf16
  __shared__ uint16_t sHA[BB * 256];       // h tile / (h0_t + h1_prev) tile
  __shared__ float    sGates[BB * WCOLS];  // f32 gate staging
  __shared__ float    sC0[BB * 32];
  __shared__ float    sC1[BB * 32];
  __shared__ float    sB0[WCOLS];
  __shared__ float    sB1[WCOLS];

  const int tid  = threadIdx.x;
  const int lane = tid & 31;
  const int wave = tid >> 5;        // N-tile (0..7)
  const int blk  = blockIdx.x >> 3; // batch block (0..3)
  const int wg   = blockIdx.x & 7;  // gate-column slice (0..7)

  // ---- startup: convert+scatter weight slices into WMMA-B LDS layout ----
  for (int idx = tid; idx < 256 * WCOLS; idx += 256) {
    int cl = idx & (WCOLS - 1);   // local gate col
    int k  = idx >> 7;            // K row
    int gcol = (cl >> 5) * 256 + wg * 32 + (cl & 31);  // global gate col
    int dest = ((k >> 5) * 8 + (cl >> 4)) * 512
             + ((((k >> 4) & 1) * 16 + (cl & 15)) * 16)
             + (((k >> 1) & 7) * 2) + (k & 1);
    sWxh[dest] = f2bf(Wxh[k * 1024 + gcol]);
    sWhh[dest] = f2bf(Whh[k * 1024 + gcol]);
  }
  if (tid < WCOLS) {
    int gcol = (tid >> 5) * 256 + wg * 32 + (tid & 31);
    sB0[tid] = bxh[gcol] + bhh[gcol];    // layer 0 combined bias
    sB1[tid] = 2.0f * bhh[gcol];         // layer 1: b_hh appears twice
  }
  sC0[tid] = 0.0f; sC0[tid + 256] = 0.0f;
  sC1[tid] = 0.0f; sC1[tid + 256] = 0.0f;
  __syncthreads();

  const float* xblk   = x   + (size_t)blk * BB * T_STEPS * DMODEL;
  float*       outblk = out + (size_t)blk * BB * T_STEPS * HDIM;
  float*       hfin   = out + (size_t)64 * T_STEPS * HDIM;
  float*       cfin   = hfin + 2 * 64 * HDIM;
  const int xrow = tid >> 4;          // 0..15
  const int xcol = (tid & 15) * 16;   // 0..240

  for (int t = 0; t < T_STEPS; ++t) {
    const int p = t & 1, q = p ^ 1;
    uint16_t* h0p = h0buf + ((size_t)(p * NBLK + blk) * BB) * HDIM;
    uint16_t* h0q = h0buf + ((size_t)(q * NBLK + blk) * BB) * HDIM;
    uint16_t* h1p = h1buf + ((size_t)(p * NBLK + blk) * BB) * HDIM;
    uint16_t* h1q = h1buf + ((size_t)(q * NBLK + blk) * BB) * HDIM;

    if (t + 1 < T_STEPS)   // prefetch next x tile -> global_prefetch_b8
      __builtin_prefetch(xblk + ((size_t)xrow * T_STEPS + (t + 1)) * DMODEL + xcol, 0, 1);

    // load x_t tile (f32->bf16) and h0_{t-1} tile into LDS
    {
      const float* xp = xblk + ((size_t)xrow * T_STEPS + t) * DMODEL + xcol;
      union { uint4 qv[2]; uint16_t h[16]; } px;
#pragma unroll
      for (int j = 0; j < 16; ++j) px.h[j] = f2bf(xp[j]);
      *(uint4*)(sXA + xrow * 256 + xcol)     = px.qv[0];
      *(uint4*)(sXA + xrow * 256 + xcol + 8) = px.qv[1];
      const uint4* hp = (const uint4*)(h0q + xrow * 256 + xcol);
      *(uint4*)(sHA + xrow * 256 + xcol)     = hp[0];
      *(uint4*)(sHA + xrow * 256 + xcol + 8) = hp[1];
    }
    __syncthreads();

    // ---- layer 0: gates0 = x_t@Wxh + h0@Whh ----
    v8f acc = {0.f, 0.f, 0.f, 0.f, 0.f, 0.f, 0.f, 0.f};
#pragma unroll
    for (int kt = 0; kt < 8; ++kt)
      acc = WMMA_BF16(load_a(sXA, lane, kt), load_b(sWxh, lane, kt, wave), acc);
#pragma unroll
    for (int kt = 0; kt < 8; ++kt)
      acc = WMMA_BF16(load_a(sHA, lane, kt), load_b(sWhh, lane, kt, wave), acc);
    {
      int n = lane & 15, mo = (lane >> 4) * 8;
#pragma unroll
      for (int r = 0; r < 8; ++r) sGates[(mo + r) * WCOLS + wave * 16 + n] = acc[r];
    }
    __syncthreads();

#pragma unroll
    for (int e2 = 0; e2 < 2; ++e2) {           // c0/h0 update: 2 elems/thread
      int e = tid * 2 + e2, r = e >> 5, u = e & 31;
      float gi = sGates[r * WCOLS + u]      + sB0[u];
      float gf = sGates[r * WCOLS + 32 + u] + sB0[32 + u];
      float gg = sGates[r * WCOLS + 64 + u] + sB0[64 + u];
      float go = sGates[r * WCOLS + 96 + u] + sB0[96 + u];
      float c  = sC0[e] * sigm(gf) + sigm(gi) * tanh_fast(gg);
      float h  = sigm(go) * tanh_fast(c);
      sC0[e] = c;
      h0p[r * HDIM + wg * 32 + u] = f2bf(h);
      if (t == T_STEPS - 1) {
        hfin[(0 * 64 + blk * BB + r) * HDIM + wg * 32 + u] = h;
        cfin[(0 * 64 + blk * BB + r) * HDIM + wg * 32 + u] = c;
      }
    }

    block_barrier(cnt + blk, 8u * (unsigned)(t + 1));   // h0_t now cluster-visible

    // ---- layer 1: A2 = h0_t + h1_{t-1} (same Whh for input & hidden) ----
    {
      union { uint4 qv[2]; uint16_t h[16]; } a0, a1, ro;
      const uint4* p0 = (const uint4*)(h0p + xrow * 256 + xcol);
      const uint4* p1 = (const uint4*)(h1q + xrow * 256 + xcol);
      a0.qv[0] = p0[0]; a0.qv[1] = p0[1];
      a1.qv[0] = p1[0]; a1.qv[1] = p1[1];
#pragma unroll
      for (int j = 0; j < 16; ++j) ro.h[j] = f2bf(bf2f(a0.h[j]) + bf2f(a1.h[j]));
      *(uint4*)(sHA + xrow * 256 + xcol)     = ro.qv[0];
      *(uint4*)(sHA + xrow * 256 + xcol + 8) = ro.qv[1];
    }
    __syncthreads();

    v8f acc2 = {0.f, 0.f, 0.f, 0.f, 0.f, 0.f, 0.f, 0.f};
#pragma unroll
    for (int kt = 0; kt < 8; ++kt)
      acc2 = WMMA_BF16(load_a(sHA, lane, kt), load_b(sWhh, lane, kt, wave), acc2);
    {
      int n = lane & 15, mo = (lane >> 4) * 8;
#pragma unroll
      for (int r = 0; r < 8; ++r) sGates[(mo + r) * WCOLS + wave * 16 + n] = acc2[r];
    }
    __syncthreads();

#pragma unroll
    for (int e2 = 0; e2 < 2; ++e2) {           // c1/h1 update + output write
      int e = tid * 2 + e2, r = e >> 5, u = e & 31;
      float gi = sGates[r * WCOLS + u]      + sB1[u];
      float gf = sGates[r * WCOLS + 32 + u] + sB1[32 + u];
      float gg = sGates[r * WCOLS + 64 + u] + sB1[64 + u];
      float go = sGates[r * WCOLS + 96 + u] + sB1[96 + u];
      float c  = sC1[e] * sigm(gf) + sigm(gi) * tanh_fast(gg);
      float h  = sigm(go) * tanh_fast(c);
      sC1[e] = c;
      h1p[r * HDIM + wg * 32 + u] = f2bf(h);
      outblk[((size_t)r * T_STEPS + t) * HDIM + wg * 32 + u] = h;
      if (t == T_STEPS - 1) {
        hfin[(1 * 64 + blk * BB + r) * HDIM + wg * 32 + u] = h;
        cfin[(1 * 64 + blk * BB + r) * HDIM + wg * 32 + u] = c;
      }
    }
    // next iteration's LDS writes are fenced by its post-load __syncthreads
  }
}

// ---------------- launcher ----------------
extern "C" void kernel_launch(void* const* d_in, const int* in_sizes, int n_in,
                              void* d_out, int out_size, void* d_ws, size_t ws_size,
                              hipStream_t stream) {
  (void)in_sizes; (void)n_in; (void)out_size; (void)ws_size;
  const float* x   = (const float*)d_in[0];
  const float* Wxh = (const float*)d_in[1];
  const float* bxh = (const float*)d_in[2];
  const float* Whh = (const float*)d_in[3];
  const float* bhh = (const float*)d_in[4];
  float* out = (float*)d_out;

  uint8_t*  ws    = (uint8_t*)d_ws;
  unsigned* cnt   = (unsigned*)ws;                       // 4 counters (256 B pad)
  uint16_t* h0buf = (uint16_t*)(ws + 256);               // [2][4][16][256] bf16
  uint16_t* h1buf = (uint16_t*)(ws + 256 + 2 * NBLK * BB * HDIM * sizeof(uint16_t));

  lstm_init<<<32, 256, 0, stream>>>(cnt, h0buf, h1buf);
  lstm_persistent<<<NBLK * NWGCOL, 256, 0, stream>>>(x, Wxh, bxh, Whh, bhh, out,
                                                     h0buf, h1buf, cnt);
}
